// SpatioTemporalSSM_12515534700923
// MI455X (gfx1250) — matrix operand order repaired and checked by reading, not compiled, not run on hardware
//
#include <hip/hip_runtime.h>
#include <hip/hip_bf16.h>
#include <math.h>

// ---------------- problem constants ----------------
constexpr int   Bn  = 4;
constexpr int   Cc  = 128;          // C == H == P
constexpr int   Lx  = 65536;        // T*H*W = 16*64*64
constexpr int   Nn  = Bn * Lx;      // 262144 tokens
constexpr int   CS  = 256;          // scan chunk size
constexpr int   NCH = Lx / CS;      // 256 chunks per batch

typedef __attribute__((ext_vector_type(8)))  float  v8f;
typedef __attribute__((ext_vector_type(8)))  __bf16 v8bf;
typedef __attribute__((ext_vector_type(16))) __bf16 v16bf;
typedef __attribute__((ext_vector_type(4)))  unsigned int u32x4;
typedef __attribute__((ext_vector_type(4)))  int          i32x4;
typedef __attribute__((ext_vector_type(8)))  int          i32x8;

#define WMMA_BF16(a, b, c) \
    __builtin_amdgcn_wmma_f32_16x16x32_bf16(false, (a), false, (b), (short)0, (c), false, false)

// ================= Tensor Data Mover (gfx1250) =================
#if defined(__has_builtin)
#  if __has_builtin(__builtin_amdgcn_tensor_load_to_lds)
#    define HAVE_TDM 1
#  endif
#endif

#ifdef HAVE_TDM
// therock-10.0 headers ship the TDM descriptor header and the 6-arg builtin;
// ROCm 7.2 (clang-22) has the 5-arg form.
#if __has_include(<hip/amd_detail/amd_gfx1250_TDM.h>)
#  define TDM_ARGS6 1
#endif
// One 2D tile: 128 rows x 128 f32, row stride `stride_elems` in memory,
// LDS destination padded +1 DWORD per 128 DWORDs (pad_interval=6, pad_amount=0)
// -> reproduces the 129-float LDS row stride in hardware.
__device__ __forceinline__ void tdm_load_tile128(void* lds_dst, const void* gsrc,
                                                 int stride_elems) {
    unsigned long long ga = (unsigned long long)(size_t)gsrc;
    u32x4 g0;
    g0[0] = 1u;                                               // count=1 (valid user D#)
    g0[1] = (unsigned int)(size_t)lds_dst;                    // lds_addr = low 32 bits of flat LDS addr
    g0[2] = (unsigned int)ga;                                 // global_addr[31:0]
    g0[3] = (unsigned int)((ga >> 32) & 0x1FFFFFFull) | (2u << 30); // addr[56:32] | type=2
    i32x8 g1;
    g1[0] = (2 << 16) | (1 << 20) | (6 << 22);                // data_size=4B, pad_en, every 128dw +1dw
    g1[1] = (int)(128u << 16);                                // tensor_dim0 = 128
    g1[2] = (int)(128u << 16);                                // tensor_dim1 = 128
    g1[3] = (int)(128u << 16);                                // tile_dim0   = 128
    g1[4] = 128;                                              // tile_dim1   = 128 (tile_dim2=0)
    g1[5] = stride_elems;                                     // tensor_dim0_stride
    g1[6] = 0; g1[7] = 0;
    i32x4 z4 = {0, 0, 0, 0};
#ifdef TDM_ARGS6
    i32x8 z8 = {0, 0, 0, 0, 0, 0, 0, 0};
    __builtin_amdgcn_tensor_load_to_lds(g0, g1, z4, z4, z8, 0);
#else
    __builtin_amdgcn_tensor_load_to_lds(g0, g1, z4, z4, 0);
#endif
}
__device__ __forceinline__ void tdm_wait() {
#if __has_builtin(__builtin_amdgcn_s_wait_tensorcnt)
    __builtin_amdgcn_s_wait_tensorcnt(0);
#else
    asm volatile("s_wait_tensorcnt 0x0" ::: "memory");
#endif
}
#endif // HAVE_TDM

// ---- fragment helpers (CDNA5 ISA 7.12.2 wave32 layouts) ----
__device__ __forceinline__ v16bf mk16(v8bf lo, v8bf hi) {
    v16bf r;
#pragma unroll
    for (int j = 0; j < 8; ++j) { r[j] = lo[j]; r[j + 8] = hi[j]; }
    return r;
}
__device__ __forceinline__ v16bf a_frag(const __bf16* rowp, int k0, int half) {
    v8bf lo = *(const v8bf*)(rowp + k0 + half * 8);
    v8bf hi = *(const v8bf*)(rowp + k0 + 16 + half * 8);
    return mk16(lo, hi);
}
__device__ __forceinline__ v16bf b_frag(const __bf16* colp, int k0, int half) {
    v8bf lo = *(const v8bf*)(colp + k0 + half * 8);
    v8bf hi = *(const v8bf*)(colp + k0 + 16 + half * 8);
    return mk16(lo, hi);
}

template <int NT> struct AccT { v8f t[2][NT]; };
template <int NT> __device__ __forceinline__ void acc_zero(AccT<NT>& a) {
#pragma unroll
    for (int i = 0; i < 2; ++i)
#pragma unroll
        for (int j = 0; j < NT; ++j)
#pragma unroll
            for (int r = 0; r < 8; ++r) a.t[i][j][r] = 0.0f;
}

// =====================================================================
// K0: per-channel SSM parameters: lam_bar, (lam_bar-1)/lam, lam_bar^CS
// =====================================================================
__global__ void k_params(const float* __restrict__ lam_re,
                         const float* __restrict__ lam_im,
                         const float* __restrict__ log_step,
                         float2* __restrict__ lamBar,
                         float2* __restrict__ acs,
                         float2* __restrict__ coef) {
    int p = threadIdx.x;
    if (p >= Cc) return;
    float dt = expf(log_step[p]);
    float ar = lam_re[p], ai = lam_im[p];
    float er  = expf(ar * dt);
    float lbr = er * cosf(ai * dt);
    float lbi = er * sinf(ai * dt);
    lamBar[p] = make_float2(lbr, lbi);
    float nr = lbr - 1.0f, ni = lbi;
    float den = ar * ar + ai * ai;
    coef[p] = make_float2((nr * ar + ni * ai) / den, (ni * ar - nr * ai) / den);
    float ecs = expf(ar * dt * (float)CS);
    acs[p] = make_float2(ecs * cosf(ai * dt * (float)CS), ecs * sinf(ai * dt * (float)CS));
}

// =====================================================================
// K1: bf16 weight prep, stored col-major-per-output (Wt[col*K + k]) so a
//     WMMA B-fragment is two contiguous 16B loads per lane.
// =====================================================================
__global__ void k_weights(const float* __restrict__ b_re, const float* __restrict__ b_im,
                          const float* __restrict__ c_re, const float* __restrict__ c_im,
                          const float* __restrict__ w1,   const float* __restrict__ w2,
                          const float2* __restrict__ coef,
                          __bf16* __restrict__ Bbar_t, __bf16* __restrict__ Ct,
                          __bf16* __restrict__ w1t,    __bf16* __restrict__ w2t) {
    int idx = blockIdx.x * blockDim.x + threadIdx.x;     // 0 .. 32767
    if (idx < Cc * Cc) {
        int p = idx >> 7, h = idx & 127;
        float2 cf = coef[p];
        float br = b_re[p * Cc + h], bi = b_im[p * Cc + h];
        Bbar_t[(size_t)p * Cc + h]        = (__bf16)(cf.x * br - cf.y * bi); // re col p
        Bbar_t[(size_t)(p + Cc) * Cc + h] = (__bf16)(cf.x * bi + cf.y * br); // im col p+128
    }
    if (idx < Cc * 2 * Cc) {
        int h = idx >> 8, j = idx & 255;
        float cv = (j < Cc) ? 2.0f * c_re[h * Cc + j] : -2.0f * c_im[h * Cc + (j - Cc)];
        Ct[(size_t)h * 256 + j]  = (__bf16)cv;                 // col=h, k=j
        w1t[(size_t)j * Cc + h]  = (__bf16)w1[h * 256 + j];    // col=j, k=h
        w2t[(size_t)h * 256 + j] = (__bf16)w2[j * Cc + h];     // col=h, k=j
    }
}

// =====================================================================
// K2: fused transpose + LN1 + GEMM1 per 128-token tile.
//     x(B,C,L) -TDM-> LDS -> h(f32) ; u = LN1 lives only in LDS
//     bu = u @ Bbar  (bf16, N x 256)
// =====================================================================
__global__ void __launch_bounds__(256)
k_stage1(const float* __restrict__ x,
         const float* __restrict__ g1, const float* __restrict__ b1n,
         const __bf16* __restrict__ Bbar_t,
         float* __restrict__ h, __bf16* __restrict__ bu) {
    __shared__ float  xt[128 * 129];   // [c][l] tile of x (129-stride, TDM pad)
    __shared__ __bf16 ut[128 * 136];   // [token][c] LN1 output
    int tid = threadIdx.x;
    int bb = blockIdx.x >> 9;
    int l0 = (blockIdx.x & 511) << 7;
    size_t row_base = (size_t)bb * Lx + l0;

#ifdef HAVE_TDM
    if (tid < 32) {                    // one wave drives the tile DMA
        tdm_load_tile128(xt, x + ((size_t)(bb * Cc)) * Lx + l0, Lx);
        tdm_wait();
    }
#else
    for (int idx = tid; idx < 128 * 128; idx += 256) {
        int c = idx >> 7, ll = idx & 127;
        xt[c * 129 + ll] = x[((size_t)(bb * Cc + c)) * Lx + l0 + ll];
    }
#endif
    __syncthreads();
    for (int idx = tid; idx < 128 * 128; idx += 256) {       // h token-major
        int r = idx >> 7, c = idx & 127;
        h[(row_base + r) * Cc + c] = xt[c * 129 + r];
    }
    int sub = tid & 7;                                       // LN1 -> ut (LDS only)
    for (int pass = 0; pass < 4; ++pass) {
        int row = pass * 32 + (tid >> 3);
        float v[16], s = 0.f, s2 = 0.f;
#pragma unroll
        for (int i = 0; i < 16; ++i) {
            float t = xt[(sub * 16 + i) * 129 + row];
            v[i] = t; s += t; s2 += t * t;
        }
#pragma unroll
        for (int mm = 1; mm < 8; mm <<= 1) { s += __shfl_xor(s, mm, 32); s2 += __shfl_xor(s2, mm, 32); }
        float mean = s * (1.0f / Cc);
        float rstd = rsqrtf(s2 * (1.0f / Cc) - mean * mean + 1e-5f);
#pragma unroll
        for (int i = 0; i < 16; ++i) {
            int c = sub * 16 + i;
            ut[row * 136 + c] = (__bf16)((v[i] - mean) * rstd * g1[c] + b1n[c]);
        }
    }
    __syncthreads();
    // GEMM: ut(128x128) @ Bbar_t(cols 256) -> bu bf16
    int lane = tid & 31, wave = tid >> 5;
    int wm = wave & 3, wn = wave >> 2;
    int m = lane & 15, half = lane >> 4;
    const __bf16* arow0 = ut + (wm * 32 + m) * 136;
    const __bf16* arow1 = ut + (wm * 32 + 16 + m) * 136;
#pragma unroll
    for (int ch = 0; ch < 2; ++ch) {
        AccT<4> acc; acc_zero(acc);
#pragma unroll
        for (int k0 = 0; k0 < 128; k0 += 32) {
            v16bf a0 = a_frag(arow0, k0, half);
            v16bf a1 = a_frag(arow1, k0, half);
#pragma unroll
            for (int j = 0; j < 4; ++j) {
                int cg = ch * 128 + wn * 64 + j * 16 + m;
                v16bf bf = b_frag(Bbar_t + (size_t)cg * 128, k0, half);
                acc.t[0][j] = WMMA_BF16(a0, bf, acc.t[0][j]);
                acc.t[1][j] = WMMA_BF16(a1, bf, acc.t[1][j]);
            }
        }
#pragma unroll
        for (int i = 0; i < 2; ++i)
#pragma unroll
            for (int j = 0; j < 4; ++j)
#pragma unroll
                for (int r = 0; r < 8; ++r) {
                    size_t ro = row_base + wm * 32 + i * 16 + r + 8 * half;
                    int    cg = ch * 128 + wn * 64 + j * 16 + m;
                    bu[ro * 256 + cg] = (__bf16)acc.t[i][j][r];
                }
    }
}

// =====================================================================
// Scan: x_t = lamBar*x_{t-1} + bu_t (complex, per (b,p)); bf16 stream,
// fp32 state; pass3 overwrites bu in place -> xs.
// =====================================================================
__global__ void k_scan1(const __bf16* __restrict__ bu, const float2* __restrict__ lamBar,
                        float2* __restrict__ carr) {
    int idx = blockIdx.x * blockDim.x + threadIdx.x;
    int p = idx & 127, ch = (idx >> 7) & (NCH - 1), b = idx >> 15;
    float2 A = lamBar[p];
    float xr = 0.f, xi = 0.f;
    const __bf16* ptr = bu + ((size_t)b * Lx + (size_t)ch * CS) * 256 + p;
    for (int t = 0; t < CS; ++t) {
        __builtin_prefetch(ptr + 16 * 256, 0, 1);
        float br = (float)ptr[0], bi = (float)ptr[128];
        float nr = fmaf(A.x, xr, fmaf(-A.y, xi, br));
        float ni = fmaf(A.x, xi, fmaf( A.y, xr, bi));
        xr = nr; xi = ni; ptr += 256;
    }
    carr[((size_t)b * NCH + ch) * Cc + p] = make_float2(xr, xi);
}

__global__ void k_scan2(const float2* __restrict__ carr, const float2* __restrict__ acs,
                        float2* __restrict__ prefix) {
    int idx = blockIdx.x * blockDim.x + threadIdx.x;
    if (idx >= Bn * Cc) return;
    int p = idx & 127, b = idx >> 7;
    float2 A = acs[p];
    float xr = 0.f, xi = 0.f;
    for (int ch = 0; ch < NCH; ++ch) {
        size_t w = ((size_t)b * NCH + ch) * Cc + p;
        prefix[w] = make_float2(xr, xi);
        float2 c = carr[w];
        float nr = fmaf(A.x, xr, fmaf(-A.y, xi, c.x));
        float ni = fmaf(A.x, xi, fmaf( A.y, xr, c.y));
        xr = nr; xi = ni;
    }
}

__global__ void k_scan3(__bf16* __restrict__ bu, const float2* __restrict__ lamBar,
                        const float2* __restrict__ prefix) {
    int idx = blockIdx.x * blockDim.x + threadIdx.x;
    int p = idx & 127, ch = (idx >> 7) & (NCH - 1), b = idx >> 15;
    float2 A = lamBar[p];
    float2 c0 = prefix[((size_t)b * NCH + ch) * Cc + p];
    float xr = c0.x, xi = c0.y;
    __bf16* ptr = bu + ((size_t)b * Lx + (size_t)ch * CS) * 256 + p;
    for (int t = 0; t < CS; ++t) {
        __builtin_prefetch(ptr + 16 * 256, 0, 1);
        float br = (float)ptr[0], bi = (float)ptr[128];
        float nr = fmaf(A.x, xr, fmaf(-A.y, xi, br));
        float ni = fmaf(A.x, xi, fmaf( A.y, xr, bi));
        xr = nr; xi = ni;
        ptr[0]   = (__bf16)xr;        // in place: bu -> xs
        ptr[128] = (__bf16)xi;
        ptr += 256;
    }
}

// =====================================================================
// K3 (tail, fully fused per 128-token tile):
//   TDM-load h tile (overlapped under the SSM-output WMMA loop)
//   h2 = h + xs@Ct + LN1(h)*d   ;  hn = LN2(h2)   (all LDS)
//   out = transpose(h2 + gelu(hn@w1+b1)@w2 + b2)
// =====================================================================
__global__ void __launch_bounds__(256)
k_tail(const __bf16* __restrict__ xs, const __bf16* __restrict__ Ct,
       const float* __restrict__ dvec,
       const float* __restrict__ g1, const float* __restrict__ b1n,
       const float* __restrict__ g2, const float* __restrict__ b2n,
       const __bf16* __restrict__ w1t, const float* __restrict__ bias1,
       const __bf16* __restrict__ w2t, const float* __restrict__ bias2,
       const float* __restrict__ h, float* __restrict__ out) {
    __shared__ float  ht[128 * 129];   // h tile -> h2 in place -> out in place
    __shared__ __bf16 hnT[128 * 136];  // LN2 output
    __shared__ __bf16 gt[128 * 136];   // one 128-wide half of gelu(hidden)
    __shared__ float  mu1[128], rs1[128];
    int tid = threadIdx.x;
    int bb = blockIdx.x >> 9;
    int l0 = (blockIdx.x & 511) << 7;
    size_t row_base = (size_t)bb * Lx + l0;
    int lane = tid & 31, wave = tid >> 5;
    int wm = wave & 3, wn = wave >> 2;
    int m = lane & 15, half = lane >> 4;

#ifdef HAVE_TDM
    if (tid < 32) tdm_load_tile128(ht, h + row_base * Cc, Cc);   // issue, don't wait yet
#else
    for (int idx = tid; idx < 128 * 128; idx += 256) {
        int r = idx >> 7, c = idx & 127;
        ht[r * 129 + c] = h[(row_base + r) * Cc + c];
    }
#endif

    // ---- SSM output GEMM (xs/Ct only — overlaps the h-tile DMA) ----
    AccT<4> accS; acc_zero(accS);
    {
        const __bf16* arow0 = xs + (row_base + wm * 32 + m) * 256;
        const __bf16* arow1 = xs + (row_base + wm * 32 + 16 + m) * 256;
#pragma unroll
        for (int k0 = 0; k0 < 256; k0 += 32) {
            v16bf a0 = a_frag(arow0, k0, half);
            v16bf a1 = a_frag(arow1, k0, half);
#pragma unroll
            for (int j = 0; j < 4; ++j) {
                int cg = wn * 64 + j * 16 + m;
                v16bf bf = b_frag(Ct + (size_t)cg * 256, k0, half);
                accS.t[0][j] = WMMA_BF16(a0, bf, accS.t[0][j]);
                accS.t[1][j] = WMMA_BF16(a1, bf, accS.t[1][j]);
            }
        }
    }
#ifdef HAVE_TDM
    if (tid < 32) tdm_wait();
#endif
    __syncthreads();                       // h tile visible to all waves

    int sub = tid & 7;                     // LN1 stats (u recompute)
    for (int pass = 0; pass < 4; ++pass) {
        int row = pass * 32 + (tid >> 3);
        float s = 0.f, s2 = 0.f;
#pragma unroll
        for (int i = 0; i < 16; ++i) {
            float t = ht[row * 129 + sub * 16 + i];
            s += t; s2 += t * t;
        }
#pragma unroll
        for (int mm = 1; mm < 8; mm <<= 1) { s += __shfl_xor(s, mm, 32); s2 += __shfl_xor(s2, mm, 32); }
        if (sub == 0) {
            float mean = s * (1.0f / Cc);
            mu1[row] = mean;
            rs1[row] = rsqrtf(s2 * (1.0f / Cc) - mean * mean + 1e-5f);
        }
    }
    __syncthreads();
    // ---- epilogue: h2 = acc + LN1(h)*d + h, in place on ht ----
#pragma unroll
    for (int i = 0; i < 2; ++i)
#pragma unroll
        for (int j = 0; j < 4; ++j)
#pragma unroll
            for (int r = 0; r < 8; ++r) {
                int ri = wm * 32 + i * 16 + r + 8 * half;
                int ci = wn * 64 + j * 16 + m;
                float hv = ht[ri * 129 + ci];
                float uv = (hv - mu1[ri]) * rs1[ri] * g1[ci] + b1n[ci];
                ht[ri * 129 + ci] = accS.t[i][j][r] + uv * dvec[ci] + hv;
            }
    __syncthreads();
    // ---- LN2 -> hnT (LDS) ----
    for (int pass = 0; pass < 4; ++pass) {
        int row = pass * 32 + (tid >> 3);
        float v[16], s = 0.f, s2 = 0.f;
#pragma unroll
        for (int i = 0; i < 16; ++i) {
            float t = ht[row * 129 + sub * 16 + i];
            v[i] = t; s += t; s2 += t * t;
        }
#pragma unroll
        for (int mm = 1; mm < 8; mm <<= 1) { s += __shfl_xor(s, mm, 32); s2 += __shfl_xor(s2, mm, 32); }
        float mean = s * (1.0f / Cc);
        float rstd = rsqrtf(s2 * (1.0f / Cc) - mean * mean + 1e-5f);
#pragma unroll
        for (int i = 0; i < 16; ++i) {
            int c = sub * 16 + i;
            hnT[row * 136 + c] = (__bf16)((v[i] - mean) * rstd * g2[c] + b2n[c]);
        }
    }
    __syncthreads();

    // ---- MLP: two hidden halves, acc2 carried across halves ----
    AccT<4> acc2; acc_zero(acc2);
    const __bf16* hrow0 = hnT + (wm * 32 + m) * 136;
    const __bf16* hrow1 = hnT + (wm * 32 + 16 + m) * 136;
    const __bf16* grow0 = gt + (wm * 32 + m) * 136;
    const __bf16* grow1 = gt + (wm * 32 + 16 + m) * 136;
    for (int chh = 0; chh < 2; ++chh) {
        AccT<4> acc1; acc_zero(acc1);
#pragma unroll
        for (int k0 = 0; k0 < 128; k0 += 32) {
            v16bf a0 = a_frag(hrow0, k0, half);
            v16bf a1 = a_frag(hrow1, k0, half);
#pragma unroll
            for (int j = 0; j < 4; ++j) {
                int cg = chh * 128 + wn * 64 + j * 16 + m;
                v16bf bf = b_frag(w1t + (size_t)cg * 128, k0, half);
                acc1.t[0][j] = WMMA_BF16(a0, bf, acc1.t[0][j]);
                acc1.t[1][j] = WMMA_BF16(a1, bf, acc1.t[1][j]);
            }
        }
        __syncthreads();                 // prior half's gt reads complete
#pragma unroll
        for (int i = 0; i < 2; ++i)
#pragma unroll
            for (int j = 0; j < 4; ++j)
#pragma unroll
                for (int r = 0; r < 8; ++r) {
                    int ri = wm * 32 + i * 16 + r + 8 * half;
                    int cl = wn * 64 + j * 16 + m;            // local col in half
                    float v = acc1.t[i][j][r] + bias1[chh * 128 + cl];
                    v = 0.5f * v * (1.0f + erff(v * 0.70710678118654752f));
                    gt[ri * 136 + cl] = (__bf16)v;
                }
        __syncthreads();                 // gt half ready
#pragma unroll
        for (int k0 = 0; k0 < 128; k0 += 32) {
            v16bf a0 = a_frag(grow0, k0, half);
            v16bf a1 = a_frag(grow1, k0, half);
#pragma unroll
            for (int j = 0; j < 4; ++j) {
                int cg = wn * 64 + j * 16 + m;
                v16bf bf = b_frag(w2t + (size_t)cg * 256 + chh * 128, k0, half);
                acc2.t[0][j] = WMMA_BF16(a0, bf, acc2.t[0][j]);
                acc2.t[1][j] = WMMA_BF16(a1, bf, acc2.t[1][j]);
            }
        }
    }
    // ---- epilogue: + bias2 + residual, in place on ht ----
#pragma unroll
    for (int i = 0; i < 2; ++i)
#pragma unroll
        for (int j = 0; j < 4; ++j)
#pragma unroll
            for (int r = 0; r < 8; ++r) {
                int ri = wm * 32 + i * 16 + r + 8 * half;
                int ci = wn * 64 + j * 16 + m;
                ht[ri * 129 + ci] = acc2.t[i][j][r] + bias2[ci] + ht[ri * 129 + ci];
            }
    __syncthreads();
    // ---- transposed store back to (B, C, L) ----
    for (int idx = tid; idx < 128 * 128; idx += 256) {
        int col = idx >> 7, r = idx & 127;
        out[((size_t)(bb * Cc + col)) * Lx + l0 + r] = ht[r * 129 + col];
    }
}

// =====================================================================
// host launcher
// =====================================================================
extern "C" void kernel_launch(void* const* d_in, const int* in_sizes, int n_in,
                              void* d_out, int out_size, void* d_ws, size_t ws_size,
                              hipStream_t stream) {
    const float* x        = (const float*)d_in[0];
    const float* n1_g     = (const float*)d_in[1];
    const float* n1_b     = (const float*)d_in[2];
    const float* lam_re   = (const float*)d_in[3];
    const float* lam_im   = (const float*)d_in[4];
    const float* b_re     = (const float*)d_in[5];
    const float* b_im     = (const float*)d_in[6];
    const float* c_re     = (const float*)d_in[7];
    const float* c_im     = (const float*)d_in[8];
    const float* dvec     = (const float*)d_in[9];
    const float* log_step = (const float*)d_in[10];
    const float* n2_g     = (const float*)d_in[11];
    const float* n2_b     = (const float*)d_in[12];
    const float* w1       = (const float*)d_in[13];
    const float* b1       = (const float*)d_in[14];
    const float* w2       = (const float*)d_in[15];
    const float* b2       = (const float*)d_in[16];
    float* out = (float*)d_out;

    char* ws = (char*)d_ws;
    size_t off = 0;
    float*  h_f    = (float*) (ws + off); off += (size_t)Nn * Cc * 4;     // 128 MiB
    __bf16* bu_bf  = (__bf16*)(ws + off); off += (size_t)Nn * 256 * 2;    // 128 MiB (bu, then xs in place)
    float2* carr   = (float2*)(ws + off); off += (size_t)Bn * NCH * Cc * 8;
    float2* prefix = (float2*)(ws + off); off += (size_t)Bn * NCH * Cc * 8;
    float2* lamBar = (float2*)(ws + off); off += Cc * 8;
    float2* acsP   = (float2*)(ws + off); off += Cc * 8;
    float2* coefP  = (float2*)(ws + off); off += Cc * 8;
    __bf16* Bbar_t = (__bf16*)(ws + off); off += (size_t)256 * Cc * 2;
    __bf16* Ct     = (__bf16*)(ws + off); off += (size_t)Cc * 256 * 2;
    __bf16* w1t    = (__bf16*)(ws + off); off += (size_t)256 * Cc * 2;
    __bf16* w2t    = (__bf16*)(ws + off); off += (size_t)Cc * 256 * 2;

    k_params<<<1, 128, 0, stream>>>(lam_re, lam_im, log_step, lamBar, acsP, coefP);
    k_weights<<<128, 256, 0, stream>>>(b_re, b_im, c_re, c_im, w1, w2, coefP,
                                       Bbar_t, Ct, w1t, w2t);
    k_stage1<<<Nn / 128, 256, 0, stream>>>(x, n1_g, n1_b, Bbar_t, h_f, bu_bf);
    k_scan1<<<(Bn * NCH * Cc) / 256, 256, 0, stream>>>(bu_bf, lamBar, carr);
    k_scan2<<<2, 256, 0, stream>>>(carr, acsP, prefix);
    k_scan3<<<(Bn * NCH * Cc) / 256, 256, 0, stream>>>(bu_bf, lamBar, prefix);
    k_tail<<<Nn / 128, 256, 0, stream>>>(bu_bf, Ct, dvec, n1_g, n1_b, n2_g, n2_b,
                                         w1t, b1, w2t, b2, h_f, out);
}
